// TransformerDecoderLayer_26972394619629
// MI455X (gfx1250) — compile-verified, matrix operand order checked by reading, not run on hardware
//
#include <hip/hip_runtime.h>
#include <hip/hip_bf16.h>
#include <math.h>
#include <stdint.h>

#define S_LEN   2048
#define H_DIM   2048
#define HEAD_DIM 128
#define N_Q     16
#define N_KV    4
#define FFH_DIM 8192
#define EPS_V   1e-5f
#define ROPE_BASE_V 10000.0f
#define CAP_V   50.0f
#define ATT_SCALE 0.08838834764831845f  // 1/sqrt(128)

typedef __attribute__((ext_vector_type(16))) __bf16 bf16x16;
typedef __attribute__((ext_vector_type(8)))  float  f32x8;
typedef __attribute__((ext_vector_type(4)))  unsigned int u32x4;
typedef __attribute__((ext_vector_type(8)))  int    i32x8;
typedef __attribute__((ext_vector_type(4)))  int    i32x4;

union Frag {
  uint4   q[2];
  bf16x16 v;
};

__device__ __forceinline__ unsigned short f2bf(float f) {
  uint32_t u = __builtin_bit_cast(uint32_t, f);
  u += 0x7FFFu + ((u >> 16) & 1u);           // round-to-nearest-even
  return (unsigned short)(u >> 16);
}

__device__ __forceinline__ uint32_t pack2bf(float lo, float hi) {
  return (uint32_t)f2bf(lo) | ((uint32_t)f2bf(hi) << 16);
}

__device__ __forceinline__ f32x8 f32x8_zero() {
  f32x8 z = {0.f, 0.f, 0.f, 0.f, 0.f, 0.f, 0.f, 0.f};
  return z;
}

__device__ __forceinline__ f32x8 wmma_bf16(const Frag& a, const Frag& b, f32x8 c) {
  return __builtin_amdgcn_wmma_f32_16x16x32_bf16(
      /*neg_a=*/false, a.v, /*neg_b=*/false, b.v,
      /*c_mod=*/(short)0, c, /*reuse_a=*/false, /*reuse_b=*/false);
}

// ---------------------------------------------------------------------------
// Tensor Data Mover: 2-D bf16 tile (tile_h rows x tile_w elems) from global
// (row stride row_stride elems) into LDS at lds_byte_off. D# per ISA 8.3/8.4.
// Caller: exactly one wave issues; that wave must s_wait_tensorcnt(0) before
// the workgroup barrier that publishes the tile.
// ---------------------------------------------------------------------------
__device__ __forceinline__ void tdm_load_2d_bf16(
    uint32_t lds_byte_off, const void* gptr,
    uint32_t tensor_w, uint32_t tensor_h, uint32_t row_stride,
    uint32_t tile_w, uint32_t tile_h) {
  uint64_t ga = (uint64_t)(uintptr_t)gptr;
  // group 0: count=1 | lds_addr | global_addr[56:0] | type=2
  u32x4 g0 = {1u,
              lds_byte_off,
              (uint32_t)(ga & 0xFFFFFFFFu),
              (uint32_t)((ga >> 32) & 0x01FFFFFFu) | (2u << 30)};
  // group 1: data_size=1(2B); tensor_dim0/1; tile_dim0/1; dim0 stride
  i32x8 g1 = {(int)(1u << 16),
              (int)((tensor_w & 0xFFFFu) << 16),
              (int)(((tensor_w >> 16) & 0xFFFFu) | ((tensor_h & 0xFFFFu) << 16)),
              (int)(((tensor_h >> 16) & 0xFFFFu) | ((tile_w & 0xFFFFu) << 16)),
              (int)(tile_h & 0xFFFFu),
              (int)row_stride,
              0, 0};
  i32x4 gz = {0, 0, 0, 0};
#if defined(__clang_major__) && __clang_major__ >= 23
  i32x8 gz8 = {0, 0, 0, 0, 0, 0, 0, 0};
  __builtin_amdgcn_tensor_load_to_lds(g0, g1, gz, gz, gz8, 0);
#else
  __builtin_amdgcn_tensor_load_to_lds(g0, g1, gz, gz, 0);
#endif
}

// ---------------------------------------------------------------------------
// RMSNorm: fp32 row -> bf16 row (x * rsqrt(mean(x^2)+eps) * w)
// ---------------------------------------------------------------------------
__global__ __launch_bounds__(256) void rmsnorm_kernel(
    const float* __restrict__ x, const float* __restrict__ w,
    unsigned short* __restrict__ out) {
  const int row = blockIdx.x;
  const int tid = threadIdx.x;
  const float* xr = x + (size_t)row * H_DIM;
  float s = 0.f;
#pragma unroll
  for (int i = 0; i < H_DIM / 256; ++i) {
    float v = xr[tid + i * 256];
    s += v * v;
  }
#pragma unroll
  for (int off = 16; off > 0; off >>= 1) s += __shfl_xor(s, off, 32);
  __shared__ float red[8];
  if ((tid & 31) == 0) red[tid >> 5] = s;
  __syncthreads();
  float tot = 0.f;
#pragma unroll
  for (int i = 0; i < 8; ++i) tot += red[i];
  const float rs = rsqrtf(tot * (1.f / (float)H_DIM) + EPS_V);
  unsigned short* o = out + (size_t)row * H_DIM;
#pragma unroll
  for (int i = 0; i < H_DIM / 256; ++i) {
    int c = tid + i * 256;
    o[c] = f2bf(xr[c] * rs * w[c]);
  }
}

// ---------------------------------------------------------------------------
// bf16 WMMA GEMM: C[M,N](f32) = A_bf16[M,K] @ W_f32->bf16[K,N]  (+ epilogues)
//   mode 0: f32 out ; mode 1: f32 out + resid ; mode 2: bf16 out = v*silu(gate)
// Block tile 128x128, k-step 32, double-buffered LDS: TDM + global W loads for
// tile k+1 issued before the 8-WMMA burst for tile k; s_wait_tensorcnt and the
// single barrier land after compute, so DMA/HBM latency hides behind WMMA.
// ---------------------------------------------------------------------------
__global__ __launch_bounds__(256) void gemm_bf16_kernel(
    const unsigned short* __restrict__ A, int lda,
    const float* __restrict__ W, int ldw,
    const float* __restrict__ resid,
    const float* __restrict__ gate,
    void* __restrict__ out, int ldc,
    int K, int mode) {
  __shared__ __align__(16) unsigned short As[2][128][32];
  __shared__ __align__(16) unsigned short Bs[2][128][32];   // stored [N][K]

  const int tid  = threadIdx.x;
  const int lane = tid & 31;
  const int wave = tid >> 5;
  const int wy = wave >> 2;            // 0..1  -> M offset wy*64
  const int wx = wave & 3;             // 0..3  -> N offset wx*32
  const int m0 = blockIdx.y * 128;
  const int n0 = blockIdx.x * 128;

  f32x8 acc[4][2];
#pragma unroll
  for (int i = 0; i < 4; ++i)
#pragma unroll
    for (int j = 0; j < 2; ++j) acc[i][j] = f32x8_zero();

  const int bk2 = (tid >> 4) * 2;      // even k: 0,2,...,30
  const int bn  = (tid & 15) * 8;      // n: 0..120 step 8
  const int mrow = lane & 15;
  const int kk   = (lane < 16) ? 0 : 8;
  const int nk   = K >> 5;

  // ---- prologue: stage tile 0 into buffer 0 ----
  if (wave == 0) {
    tdm_load_2d_bf16((uint32_t)(uintptr_t)&As[0][0][0],
                     A + (size_t)m0 * lda,
                     (uint32_t)K, 128, (uint32_t)lda, 32, 128);
  }
  {
    const float* wp0 = W + (size_t)bk2 * ldw + n0 + bn;
    float4 a0 = *(const float4*)(wp0);
    float4 a1 = *(const float4*)(wp0 + 4);
    float4 b0 = *(const float4*)(wp0 + ldw);
    float4 b1 = *(const float4*)(wp0 + ldw + 4);
    *(uint32_t*)&Bs[0][bn + 0][bk2] = pack2bf(a0.x, b0.x);
    *(uint32_t*)&Bs[0][bn + 1][bk2] = pack2bf(a0.y, b0.y);
    *(uint32_t*)&Bs[0][bn + 2][bk2] = pack2bf(a0.z, b0.z);
    *(uint32_t*)&Bs[0][bn + 3][bk2] = pack2bf(a0.w, b0.w);
    *(uint32_t*)&Bs[0][bn + 4][bk2] = pack2bf(a1.x, b1.x);
    *(uint32_t*)&Bs[0][bn + 5][bk2] = pack2bf(a1.y, b1.y);
    *(uint32_t*)&Bs[0][bn + 6][bk2] = pack2bf(a1.z, b1.z);
    *(uint32_t*)&Bs[0][bn + 7][bk2] = pack2bf(a1.w, b1.w);
  }
  if (wave == 0) __builtin_amdgcn_s_wait_tensorcnt((short)0);
  __syncthreads();

  for (int kt = 0; kt < nk; ++kt) {
    const int cur = kt & 1;
    const int nxt = cur ^ 1;
    const int k1 = (kt + 1) << 5;
    const bool hasnext = (kt + 1) < nk;

    // ---- issue next tile's DMA + HBM loads before computing current ----
    if (hasnext && wave == 0) {
      tdm_load_2d_bf16((uint32_t)(uintptr_t)&As[nxt][0][0],
                       A + (size_t)m0 * lda + k1,
                       (uint32_t)K, 128, (uint32_t)lda, 32, 128);
    }
    float4 a0, a1, b0, b1;
    if (hasnext) {
      const float* wp0 = W + (size_t)(k1 + bk2) * ldw + n0 + bn;
      a0 = *(const float4*)(wp0);
      a1 = *(const float4*)(wp0 + 4);
      b0 = *(const float4*)(wp0 + ldw);
      b1 = *(const float4*)(wp0 + ldw + 4);
      if (k1 + 32 < K)  // pull tile k+2 toward L2 while k computes
        __builtin_prefetch(W + (size_t)(k1 + 32 + (bk2 >> 1)) * ldw + n0 + bn,
                           0, 1);
    }

    // ---- compute current tile: 8 WMMA / wave ----
    Frag af[4], bf[2];
#pragma unroll
    for (int mt = 0; mt < 4; ++mt) {
      const unsigned short* p = &As[cur][wy * 64 + mt * 16 + mrow][kk];
      af[mt].q[0] = *(const uint4*)p;
      af[mt].q[1] = *(const uint4*)(p + 16);
    }
#pragma unroll
    for (int nt = 0; nt < 2; ++nt) {
      const unsigned short* p = &Bs[cur][wx * 32 + nt * 16 + mrow][kk];
      bf[nt].q[0] = *(const uint4*)p;
      bf[nt].q[1] = *(const uint4*)(p + 16);
    }
#pragma unroll
    for (int mt = 0; mt < 4; ++mt)
#pragma unroll
      for (int nt = 0; nt < 2; ++nt)
        acc[mt][nt] = wmma_bf16(af[mt], bf[nt], acc[mt][nt]);

    // ---- publish next buffer ----
    if (hasnext) {
      *(uint32_t*)&Bs[nxt][bn + 0][bk2] = pack2bf(a0.x, b0.x);
      *(uint32_t*)&Bs[nxt][bn + 1][bk2] = pack2bf(a0.y, b0.y);
      *(uint32_t*)&Bs[nxt][bn + 2][bk2] = pack2bf(a0.z, b0.z);
      *(uint32_t*)&Bs[nxt][bn + 3][bk2] = pack2bf(a0.w, b0.w);
      *(uint32_t*)&Bs[nxt][bn + 4][bk2] = pack2bf(a1.x, b1.x);
      *(uint32_t*)&Bs[nxt][bn + 5][bk2] = pack2bf(a1.y, b1.y);
      *(uint32_t*)&Bs[nxt][bn + 6][bk2] = pack2bf(a1.z, b1.z);
      *(uint32_t*)&Bs[nxt][bn + 7][bk2] = pack2bf(a1.w, b1.w);
      if (wave == 0) __builtin_amdgcn_s_wait_tensorcnt((short)0);
    }
    __syncthreads();
  }

  const int rowoff = (lane >= 16) ? 8 : 0;
#pragma unroll
  for (int mt = 0; mt < 4; ++mt) {
#pragma unroll
    for (int nt = 0; nt < 2; ++nt) {
#pragma unroll
      for (int r = 0; r < 8; ++r) {
        int grow = m0 + wy * 64 + mt * 16 + r + rowoff;
        int gcol = n0 + wx * 32 + nt * 16 + mrow;
        size_t idx = (size_t)grow * ldc + gcol;
        float v = acc[mt][nt][r];
        if (mode == 1) v += resid[idx];
        if (mode == 2) {
          float g = gate[idx];
          v *= g / (1.f + __expf(-g));              // silu(gate) * up
          ((unsigned short*)out)[idx] = f2bf(v);
        } else {
          ((float*)out)[idx] = v;
        }
      }
    }
  }
}

// ---------------------------------------------------------------------------
// RoPE + qk-RMSNorm (+attn scale folded into q) + bf16 convert; v: bf16 only.
// grid (S, 24): h<16 -> q head, 16..19 -> k head, 20..23 -> v head. 64 thr.
// ---------------------------------------------------------------------------
__global__ __launch_bounds__(64) void rope_qknorm_kernel(
    const float* __restrict__ qf, const float* __restrict__ kf,
    const float* __restrict__ vf,
    unsigned short* __restrict__ qb, unsigned short* __restrict__ kb,
    unsigned short* __restrict__ vb) {
  const int s = blockIdx.x;
  const int h = blockIdx.y;
  const int t = threadIdx.x;  // 0..63 handles dims t and t+64
  const float* src;
  unsigned short* dst;
  bool isq = false, dorope = true;
  if (h < N_Q) {
    src = qf + ((size_t)s * N_Q + h) * HEAD_DIM;
    dst = qb + ((size_t)s * N_Q + h) * HEAD_DIM;
    isq = true;
  } else if (h < N_Q + N_KV) {
    int hk = h - N_Q;
    src = kf + ((size_t)s * N_KV + hk) * HEAD_DIM;
    dst = kb + ((size_t)s * N_KV + hk) * HEAD_DIM;
  } else {
    int hv = h - N_Q - N_KV;
    src = vf + ((size_t)s * N_KV + hv) * HEAD_DIM;
    dst = vb + ((size_t)s * N_KV + hv) * HEAD_DIM;
    dorope = false;
  }
  float xa = src[t], xb = src[t + 64];
  float ra = xa, rb = xb;
  float rs = 1.f;
  if (dorope) {
    float invf = __powf(ROPE_BASE_V, -(float)t * (1.f / 64.f));
    float ang = (float)s * invf;
    float c, sn;
    __sincosf(ang, &c, &sn);
    ra = xa * c - xb * sn;   // d < 64
    rb = xb * c + xa * sn;   // d >= 64 (same angle)
    float ss = ra * ra + rb * rb;
#pragma unroll
    for (int off = 16; off > 0; off >>= 1) ss += __shfl_xor(ss, off, 32);
    __shared__ float red2[2];
    if ((t & 31) == 0) red2[t >> 5] = ss;
    __syncthreads();
    float tot = red2[0] + red2[1];
    rs = rsqrtf(tot * (1.f / (float)HEAD_DIM) + EPS_V);
    if (isq) rs *= ATT_SCALE;
  }
  dst[t]      = f2bf(ra * rs);
  dst[t + 64] = f2bf(rb * rs);
}

// ---------------------------------------------------------------------------
// Flash attention (causal, GQA 16q/4kv, tanh softcap; clip(0,1) is identity).
// grid (S/128, N_Q); 8 waves, each wave owns a 16-row query tile.
// K chunk staged by TDM; V chunk transposed by VALU lanes; 16 WMMA / chunk.
// ---------------------------------------------------------------------------
__global__ __launch_bounds__(256) void attn_kernel(
    const unsigned short* __restrict__ qb, const unsigned short* __restrict__ kb,
    const unsigned short* __restrict__ vb, unsigned short* __restrict__ ab) {
  __shared__ __align__(16) unsigned short Ks[32][128];       // [key][hd]
  __shared__ __align__(16) unsigned short Vt[128][32];       // [hd][key]
  __shared__ __align__(16) unsigned short Ps[8][16][32];     // per-wave P tile

  const int tid   = threadIdx.x;
  const int lane  = tid & 31;
  const int wave  = tid >> 5;
  const int hq    = blockIdx.y;
  const int hkv   = hq >> 2;
  const int qrow0 = blockIdx.x * 128 + wave * 16;
  const int mrow  = lane & 15;
  const int kk    = (lane < 16) ? 0 : 8;
  const int rowoff = (lane >= 16) ? 8 : 0;
  const uint32_t ks_lds = (uint32_t)(uintptr_t)&Ks[0][0];

  // Preload Q fragments (A-operand layout), 1/sqrt(d) already folded in.
  Frag qa[4];
  {
    const unsigned short* qp =
        qb + ((size_t)(qrow0 + mrow) * N_Q + hq) * HEAD_DIM;
#pragma unroll
    for (int hk = 0; hk < 4; ++hk) {
      qa[hk].q[0] = *(const uint4*)(qp + hk * 32 + kk);
      qa[hk].q[1] = *(const uint4*)(qp + hk * 32 + kk + 16);
    }
  }

  f32x8 o[8];
  float mS[8], lS[8];
#pragma unroll
  for (int i = 0; i < 8; ++i) {
    o[i] = f32x8_zero();
    mS[i] = -INFINITY;
    lS[i] = 0.f;
  }

  const int nchunks = blockIdx.x * 4 + 4;  // keys 0 .. blockMaxRow
  for (int c = 0; c < nchunks; ++c) {
    const int kbase = c * 32;
    __syncthreads();
    // K chunk 32x128 via TDM (strided rows, no conversion needed)
    if (wave == 0) {
      tdm_load_2d_bf16(ks_lds,
                       kb + ((size_t)kbase * N_KV + hkv) * HEAD_DIM,
                       /*tensor_w=*/HEAD_DIM, /*tensor_h=*/S_LEN * N_KV,
                       /*row_stride=*/N_KV * HEAD_DIM,
                       /*tile_w=*/HEAD_DIM, /*tile_h=*/32);
    }
    {  // V chunk transposed into LDS by all threads
      int key = tid >> 3;
      int seg = (tid & 7) * 16;
      const unsigned short* vp =
          vb + ((size_t)(kbase + key) * N_KV + hkv) * HEAD_DIM + seg;
      union { uint4 q[2]; unsigned short u[16]; } vt;
      vt.q[0] = *(const uint4*)vp;
      vt.q[1] = *(const uint4*)(vp + 8);
#pragma unroll
      for (int j = 0; j < 16; ++j) Vt[seg + j][key] = vt.u[j];
    }
    if (wave == 0) __builtin_amdgcn_s_wait_tensorcnt((short)0);
    __syncthreads();

    if (kbase <= qrow0 + 15) {
      // S = Q @ K^T  (two 16-col tiles, 4 hd k-steps)
      f32x8 s0 = f32x8_zero(), s1 = f32x8_zero();
#pragma unroll
      for (int hk = 0; hk < 4; ++hk) {
        Frag b0, b1;
        const unsigned short* p0 = &Ks[mrow][hk * 32 + kk];
        const unsigned short* p1 = &Ks[mrow + 16][hk * 32 + kk];
        b0.q[0] = *(const uint4*)p0; b0.q[1] = *(const uint4*)(p0 + 16);
        b1.q[0] = *(const uint4*)p1; b1.q[1] = *(const uint4*)(p1 + 16);
        s0 = wmma_bf16(qa[hk], b0, s0);
        s1 = wmma_bf16(qa[hk], b1, s1);
      }
      // softcap + causal mask + online softmax (C-layout rows)
#pragma unroll
      for (int r = 0; r < 8; ++r) {
        int row  = qrow0 + r + rowoff;
        int col0 = kbase + mrow;
        float v0 = CAP_V * tanhf(s0[r] * (1.f / CAP_V));
        float v1 = CAP_V * tanhf(s1[r] * (1.f / CAP_V));
        if (col0 > row)      v0 = -INFINITY;
        if (col0 + 16 > row) v1 = -INFINITY;
        float rv = fmaxf(v0, v1);
#pragma unroll
        for (int off = 1; off < 16; off <<= 1)
          rv = fmaxf(rv, __shfl_xor(rv, off, 32));
        float mnew = fmaxf(mS[r], rv);
        float corr = __expf(mS[r] - mnew);
        float p0 = __expf(v0 - mnew);
        float p1 = __expf(v1 - mnew);
        float ps = p0 + p1;
#pragma unroll
        for (int off = 1; off < 16; off <<= 1) ps += __shfl_xor(ps, off, 32);
        lS[r] = lS[r] * corr + ps;
        mS[r] = mnew;
        Ps[wave][r + rowoff][mrow]      = f2bf(p0);
        Ps[wave][r + rowoff][mrow + 16] = f2bf(p1);
#pragma unroll
        for (int nt = 0; nt < 8; ++nt) o[nt][r] *= corr;
      }
      // O += P @ V  (LDS round-trip re-layouts P into A-operand pattern;
      // per-wave DS ops are in-order)
      Frag pa;
      {
        const unsigned short* pp = &Ps[wave][mrow][kk];
        pa.q[0] = *(const uint4*)pp;
        pa.q[1] = *(const uint4*)(pp + 16);
      }
#pragma unroll
      for (int nt = 0; nt < 8; ++nt) {
        Frag vfr;
        const unsigned short* vp2 = &Vt[nt * 16 + mrow][kk];
        vfr.q[0] = *(const uint4*)vp2;
        vfr.q[1] = *(const uint4*)(vp2 + 16);
        o[nt] = wmma_bf16(pa, vfr, o[nt]);
      }
    }
  }
  // epilogue: normalize and store bf16 into [row][hq*128 + hd]
#pragma unroll
  for (int r = 0; r < 8; ++r) {
    float inv = 1.f / lS[r];
    size_t row = (size_t)(qrow0 + r + rowoff);
#pragma unroll
    for (int nt = 0; nt < 8; ++nt) {
      int col = hq * HEAD_DIM + nt * 16 + mrow;
      ab[row * (N_Q * HEAD_DIM) + col] = f2bf(o[nt][r] * inv);
    }
  }
}

// ---------------------------------------------------------------------------
extern "C" void kernel_launch(void* const* d_in, const int* in_sizes, int n_in,
                              void* d_out, int out_size, void* d_ws,
                              size_t ws_size, hipStream_t stream) {
  (void)in_sizes; (void)n_in; (void)out_size; (void)ws_size;
  const float* x           = (const float*)d_in[0];
  const float* attn_norm_w = (const float*)d_in[1];
  const float* mlp_norm_w  = (const float*)d_in[2];
  const float* wq          = (const float*)d_in[3];
  const float* wk          = (const float*)d_in[4];
  const float* wv          = (const float*)d_in[5];
  const float* wo          = (const float*)d_in[6];
  const float* w_gate      = (const float*)d_in[7];
  const float* w_up        = (const float*)d_in[8];
  const float* w_down      = (const float*)d_in[9];

  char* ws = (char*)d_ws;
  size_t off = 0;
  float* x1 = (float*)(ws + off);            off += (size_t)S_LEN * H_DIM * 4;
  unsigned short* h2b = (unsigned short*)(ws + off); off += (size_t)S_LEN * H_DIM * 2;
  const size_t pbase = off;  // phase-1 buffers (dead before MLP) start here
  unsigned short* hb = (unsigned short*)(ws + off);  off += (size_t)S_LEN * H_DIM * 2;
  float* qf = (float*)(ws + off);            off += (size_t)S_LEN * N_Q  * HEAD_DIM * 4;
  float* kf = (float*)(ws + off);            off += (size_t)S_LEN * N_KV * HEAD_DIM * 4;
  float* vf = (float*)(ws + off);            off += (size_t)S_LEN * N_KV * HEAD_DIM * 4;
  unsigned short* qbp = (unsigned short*)(ws + off); off += (size_t)S_LEN * N_Q  * HEAD_DIM * 2;
  unsigned short* kbp = (unsigned short*)(ws + off); off += (size_t)S_LEN * N_KV * HEAD_DIM * 2;
  unsigned short* vbp = (unsigned short*)(ws + off); off += (size_t)S_LEN * N_KV * HEAD_DIM * 2;
  unsigned short* ab  = (unsigned short*)(ws + off); off += (size_t)S_LEN * N_Q  * HEAD_DIM * 2;
  // phase-2 overlays phase-1
  float* gf = (float*)(ws + pbase);
  unsigned short* fb = (unsigned short*)(ws + pbase + (size_t)S_LEN * FFH_DIM * 4);

  // 1. attn RMSNorm -> bf16
  rmsnorm_kernel<<<S_LEN, 256, 0, stream>>>(x, attn_norm_w, hb);
  // 2-4. Q/K/V projections
  gemm_bf16_kernel<<<dim3((N_Q * HEAD_DIM) / 128, S_LEN / 128), 256, 0, stream>>>(
      hb, H_DIM, wq, N_Q * HEAD_DIM, nullptr, nullptr, qf, N_Q * HEAD_DIM,
      H_DIM, 0);
  gemm_bf16_kernel<<<dim3((N_KV * HEAD_DIM) / 128, S_LEN / 128), 256, 0, stream>>>(
      hb, H_DIM, wk, N_KV * HEAD_DIM, nullptr, nullptr, kf, N_KV * HEAD_DIM,
      H_DIM, 0);
  gemm_bf16_kernel<<<dim3((N_KV * HEAD_DIM) / 128, S_LEN / 128), 256, 0, stream>>>(
      hb, H_DIM, wv, N_KV * HEAD_DIM, nullptr, nullptr, vf, N_KV * HEAD_DIM,
      H_DIM, 0);
  // 5. RoPE + qk-norm + bf16
  rope_qknorm_kernel<<<dim3(S_LEN, N_Q + 2 * N_KV), 64, 0, stream>>>(
      qf, kf, vf, qbp, kbp, vbp);
  // 6. flash attention
  attn_kernel<<<dim3(S_LEN / 128, N_Q), 256, 0, stream>>>(qbp, kbp, vbp, ab);
  // 7. output projection + residual -> x1
  gemm_bf16_kernel<<<dim3(H_DIM / 128, S_LEN / 128), 256, 0, stream>>>(
      ab, N_Q * HEAD_DIM, wo, H_DIM, x, nullptr, x1, H_DIM, N_Q * HEAD_DIM, 1);
  // 8. MLP RMSNorm -> bf16
  rmsnorm_kernel<<<S_LEN, 256, 0, stream>>>(x1, mlp_norm_w, h2b);
  // 9. gate proj (f32)
  gemm_bf16_kernel<<<dim3(FFH_DIM / 128, S_LEN / 128), 256, 0, stream>>>(
      h2b, H_DIM, w_gate, FFH_DIM, nullptr, nullptr, gf, FFH_DIM, H_DIM, 0);
  // 10. up proj fused with silu(gate)*up -> bf16
  gemm_bf16_kernel<<<dim3(FFH_DIM / 128, S_LEN / 128), 256, 0, stream>>>(
      h2b, H_DIM, w_up, FFH_DIM, nullptr, gf, fb, FFH_DIM, H_DIM, 2);
  // 11. down proj + residual -> d_out
  gemm_bf16_kernel<<<dim3(H_DIM / 128, S_LEN / 128), 256, 0, stream>>>(
      fb, FFH_DIM, w_down, H_DIM, x1, nullptr, d_out, H_DIM, FFH_DIM, 1);
}
